// MultiHeadAttention_82532091560485
// MI455X (gfx1250) — matrix operand order repaired and checked by reading, not compiled, not run on hardware
//
#include <hip/hip_runtime.h>
#include <hip/hip_bf16.h>

// ---------------------------------------------------------------------------
// MHA forward for MI455X (gfx1250, wave32, WMMA).
//   k_wt    : W fp32[in][out] -> Wt f16[out][in] (B-operand friendly)
//   k_proj  : fused f32->f16 projection GEMM (WMMA), per-head output,
//             LDS-staged coalesced b128 stores (templated Q/K vs V^T layout)
//   k_attn  : flash attention, 32 queries/wave, WMMA QK^T + online softmax
//             + WMMA P.V, P relaid through per-wave LDS tile
//   k_out   : output projection GEMM (WMMA) -> fp32, LDS-staged stores
// Roofline: ~90 GFLOP total, ~110MB HBM floor -> compute bound; f16 WMMA
// with f32 accumulation; all intermediates (~55MB) live in the 192MB L2.
// ---------------------------------------------------------------------------

#define DMODEL 768
#define NHEADS 12
#define DK     64
#define SEQ    2048
#define BATCH  4
#define MTOT   (BATCH * SEQ)   // 8192 rows

typedef _Float16 f16;
typedef __attribute__((ext_vector_type(16))) _Float16 v16h;
typedef __attribute__((ext_vector_type(8)))  _Float16 v8h;
typedef __attribute__((ext_vector_type(8)))  float    v8f;
typedef __attribute__((ext_vector_type(4)))  float    v4f;

// wave-private LDS store->load ordering (DS ops are in-order per wave)
#define STAGE_SYNC() asm volatile("s_wait_dscnt 0" ::: "memory")

// D = A(16x32 f16) x B(32x16 f16) + C(16x16 f32)
static __device__ __forceinline__ v8f wmma32(v16h a, v16h b, v8f c) {
    return __builtin_amdgcn_wmma_f32_16x16x32_f16(
        false, a, false, b, (short)0, c, false, false);
}

// A operand (16x32, row-major f16 source): lanes 0-15 hold row M=lane,
// K chunks {0..7,16..23}; lanes 16-31 same rows, K chunks {8..15,24..31}.
static __device__ __forceinline__ v16h load_a_f16(const f16* __restrict__ base,
                                                  int ld, int m0, int k0, int lane) {
    int r = lane & 15, hi = lane >> 4;
    const f16* p = base + (size_t)(m0 + r) * ld + k0;
    v8h h0 = *(const v8h*)(p + hi * 8);
    v8h h1 = *(const v8h*)(p + 16 + hi * 8);
    return __builtin_shufflevector(h0, h1, 0,1,2,3,4,5,6,7,8,9,10,11,12,13,14,15);
}

// Same but fp32 source with fused downconvert (projection GEMMs).
static __device__ __forceinline__ v16h load_a_f32(const float* __restrict__ base,
                                                  int ld, int m0, int k0, int lane) {
    int r = lane & 15, hi = lane >> 4;
    const float* p = base + (size_t)(m0 + r) * ld + k0;
    v8f c0 = *(const v8f*)(p + hi * 8);
    v8f c1 = *(const v8f*)(p + 16 + hi * 8);
    v8h h0 = __builtin_convertvector(c0, v8h);
    v8h h1 = __builtin_convertvector(c1, v8h);
    return __builtin_shufflevector(h0, h1, 0,1,2,3,4,5,6,7,8,9,10,11,12,13,14,15);
}

// B operand (32x16) from B^T stored row-major: lane c -> col n0+c, K chunks
// {0..7,16..23}; lanes 16-31 -> K chunks {8..15,24..31}. Two 16B loads per lane.
static __device__ __forceinline__ v16h load_bt_f16(const f16* __restrict__ bt,
                                                   int ld, int n0, int k0, int lane) {
    int c = lane & 15, hi = lane >> 4;
    const f16* p = bt + (size_t)(n0 + c) * ld + k0;
    v8h h0 = *(const v8h*)(p + hi * 8);
    v8h h1 = *(const v8h*)(p + 16 + hi * 8);
    return __builtin_shufflevector(h0, h1, 0,1,2,3,4,5,6,7,8,9,10,11,12,13,14,15);
}

// ---------------------------------------------------------------------------
__global__ void k_wt(const float* __restrict__ W, f16* __restrict__ Wt) {
    int idx = blockIdx.x * blockDim.x + threadIdx.x;
    if (idx >= DMODEL * DMODEL) return;
    int o = idx / DMODEL;
    int i = idx - o * DMODEL;
    Wt[(size_t)o * DMODEL + i] = (f16)W[(size_t)i * DMODEL + o];
}

// ---------------------------------------------------------------------------
// Projection GEMM: one wave -> 16(M) x 64(N) strip (= one head slice).
// VMODE==0: out[bh][s][64] (Q,K).  VMODE==1: out[bh][64][s] (V^T).
// Epilogue staged through per-wave LDS, written as coalesced b128 rows.
// ---------------------------------------------------------------------------
template <int VMODE>
__global__ void __launch_bounds__(128) k_proj(const float* __restrict__ X,
                                              const f16* __restrict__ Wt,
                                              const float* __restrict__ bias,
                                              f16* __restrict__ out) {
    __shared__ f16 sbuf[4][16 * 64];     // 2KB per wave
    int lane = threadIdx.x & 31;
    int wave = threadIdx.x >> 5;
    int t  = blockIdx.x * 4 + wave;      // 0 .. 512*12-1
    int tm = t / NHEADS;
    int h  = t - tm * NHEADS;
    int m0 = tm * 16;
    int n0 = h * DK;
    int col = lane & 15, hi = lane >> 4;

    v8f zero = {};
    v8f acc[4] = {zero, zero, zero, zero};
    for (int k0 = 0; k0 < DMODEL; k0 += 32) {
        if (k0 + 64 < DMODEL) {          // prefetch 2 k-steps ahead
            __builtin_prefetch(X + (size_t)(m0 + col) * DMODEL + k0 + 64 + hi * 8, 0, 3);
            __builtin_prefetch(Wt + (size_t)(n0 + col) * DMODEL + k0 + 64 + hi * 16, 0, 3);
        }
        v16h a = load_a_f32(X, DMODEL, m0, k0, lane);
        #pragma unroll
        for (int dt = 0; dt < 4; ++dt) {
            v16h b = load_bt_f16(Wt, DMODEL, n0 + dt * 16, k0, lane);
            acc[dt] = wmma32(a, b, acc[dt]);
        }
    }

    int b  = m0 / SEQ;                   // 16-row tile never crosses batch
    int s0 = m0 - b * SEQ;
    int bh = b * NHEADS + h;
    f16* S = sbuf[wave];
    #pragma unroll
    for (int dt = 0; dt < 4; ++dt) {
        float bb = bias[n0 + dt * 16 + col];
        #pragma unroll
        for (int r = 0; r < 8; ++r) {
            f16 val = (f16)(acc[dt][r] + bb);
            int row = r + 8 * hi;
            if (VMODE) S[(dt * 16 + col) * 16 + row] = val;   // [d][s] tile
            else       S[row * 64 + dt * 16 + col]   = val;   // [s][d] tile
        }
    }
    STAGE_SYNC();
    #pragma unroll
    for (int it = 0; it < 4; ++it) {     // 128 x 16B chunks, coalesced
        int chunk = it * 32 + lane;
        if (VMODE) {
            int d = chunk >> 1, off = chunk & 1;                // 64 rows x 32B
            v8h val = *(const v8h*)(S + d * 16 + off * 8);
            *(v8h*)(out + ((size_t)bh * DK + d) * SEQ + s0 + off * 8) = val;
        } else {
            int row = chunk >> 3, off = chunk & 7;              // 16 rows x 128B
            v8h val = *(const v8h*)(S + row * 64 + off * 8);
            *(v8h*)(out + ((size_t)bh * SEQ + s0 + row) * DK + off * 8) = val;
        }
    }
}

// ---------------------------------------------------------------------------
// Flash attention: one wave per (b,h, 32-query tile). Per 32-key step:
// 8 WMMAs for S = Q K^T (K tiles reused across both M-tiles), online softmax,
// P via per-wave LDS into A layout, 8 WMMAs for O += P V (V reused).
// ---------------------------------------------------------------------------
__global__ void __launch_bounds__(128) k_attn(const f16* __restrict__ Qh,
                                              const f16* __restrict__ Kh,
                                              const f16* __restrict__ Vt,
                                              f16* __restrict__ Xa) {
    __shared__ f16 pbuf[4][32 * 32];     // per-wave P tile (32q x 32k), 2KB
    __shared__ f16 obuf[4][32 * 64];     // per-wave O staging, 4KB
    int lane = threadIdx.x & 31;
    int wave = threadIdx.x >> 5;
    int w  = blockIdx.x * 4 + wave;      // 0..3071
    int qt = w & 63;                     // 64 q-tiles of 32
    int bh = w >> 6;                     // 0..47
    int m0 = qt * 32;
    int col = lane & 15, hi = lane >> 4;

    const f16* Q = Qh + (size_t)bh * SEQ * DK;
    const f16* K = Kh + (size_t)bh * SEQ * DK;
    const f16* V = Vt + (size_t)bh * DK * SEQ;

    v16h qa[2][2];
    #pragma unroll
    for (int mt = 0; mt < 2; ++mt) {
        qa[mt][0] = load_a_f16(Q, DK, m0 + mt * 16, 0,  lane);
        qa[mt][1] = load_a_f16(Q, DK, m0 + mt * 16, 32, lane);
    }

    v8f zero = {};
    v8f acc[2][4];
    float mi[2][8], li[2][8];
    #pragma unroll
    for (int mt = 0; mt < 2; ++mt) {
        #pragma unroll
        for (int dt = 0; dt < 4; ++dt) acc[mt][dt] = zero;
        #pragma unroll
        for (int r = 0; r < 8; ++r) { mi[mt][r] = -1e30f; li[mt][r] = 0.0f; }
    }

    f16* P = pbuf[wave];
    const float scale = 0.125f;          // 1/sqrt(d_k)

    for (int kc = 0; kc < SEQ; kc += 32) {
        if (kc + 64 < SEQ) {             // prefetch next-next K/V tiles
            __builtin_prefetch(K + (size_t)(kc + 64 + lane) * DK, 0, 3);
            __builtin_prefetch(V + (size_t)lane * SEQ + kc + 64, 0, 3);
            __builtin_prefetch(V + (size_t)(lane + 32) * SEQ + kc + 64, 0, 3);
        }
        v16h kb[2][2];
        #pragma unroll
        for (int nt = 0; nt < 2; ++nt) {
            kb[nt][0] = load_bt_f16(K, DK, kc + nt * 16, 0,  lane);
            kb[nt][1] = load_bt_f16(K, DK, kc + nt * 16, 32, lane);
        }
        v8f sc[2][2];
        #pragma unroll
        for (int mt = 0; mt < 2; ++mt) {
            #pragma unroll
            for (int nt = 0; nt < 2; ++nt) {
                v8f s = zero;
                s = wmma32(qa[mt][0], kb[nt][0], s);
                s = wmma32(qa[mt][1], kb[nt][1], s);
                sc[mt][nt] = s;
            }
        }

        #pragma unroll
        for (int mt = 0; mt < 2; ++mt) {
            float corr[8];
            #pragma unroll
            for (int r = 0; r < 8; ++r) {
                float t0 = sc[mt][0][r] * scale;
                float t1 = sc[mt][1][r] * scale;
                float mx = fmaxf(t0, t1);
                #pragma unroll
                for (int msk = 1; msk <= 8; msk <<= 1)
                    mx = fmaxf(mx, __shfl_xor(mx, msk, 32));
                float mnew = fmaxf(mi[mt][r], mx);
                float c  = __expf(mi[mt][r] - mnew);
                float e0 = __expf(t0 - mnew);
                float e1 = __expf(t1 - mnew);
                float rs = e0 + e1;
                #pragma unroll
                for (int msk = 1; msk <= 8; msk <<= 1)
                    rs += __shfl_xor(rs, msk, 32);
                li[mt][r] = li[mt][r] * c + rs;
                mi[mt][r] = mnew;
                corr[r]   = c;
                int row = mt * 16 + r + 8 * hi;
                P[row * 32 + col]      = (f16)e0;
                P[row * 32 + col + 16] = (f16)e1;
            }
            #pragma unroll
            for (int dt = 0; dt < 4; ++dt) {
                #pragma unroll
                for (int r = 0; r < 8; ++r) acc[mt][dt][r] *= corr[r];
            }
        }

        STAGE_SYNC();                    // P stores -> P loads (same wave)

        v16h pa[2];
        #pragma unroll
        for (int mt = 0; mt < 2; ++mt) {
            const f16* pp = P + (size_t)(mt * 16 + col) * 32;
            v8h h0 = *(const v8h*)(pp + hi * 8);
            v8h h1 = *(const v8h*)(pp + 16 + hi * 8);
            pa[mt] = __builtin_shufflevector(h0, h1,
                     0,1,2,3,4,5,6,7,8,9,10,11,12,13,14,15);
        }
        #pragma unroll
        for (int dt = 0; dt < 4; ++dt) {
            v16h vb = load_bt_f16(V, SEQ, dt * 16, kc, lane);
            acc[0][dt] = wmma32(pa[0], vb, acc[0][dt]);
            acc[1][dt] = wmma32(pa[1], vb, acc[1][dt]);
        }
    }

    // epilogue: normalize, stage 32x64 f16 tile in LDS, coalesced b128 out
    f16* O = obuf[wave];
    #pragma unroll
    for (int mt = 0; mt < 2; ++mt) {
        #pragma unroll
        for (int r = 0; r < 8; ++r) {
            float inv = 1.0f / li[mt][r];
            int row = mt * 16 + r + 8 * hi;
            #pragma unroll
            for (int dt = 0; dt < 4; ++dt)
                O[row * 64 + dt * 16 + col] = (f16)(acc[mt][dt][r] * inv);
        }
    }
    STAGE_SYNC();
    int b = bh / NHEADS, h = bh % NHEADS;
    size_t rowbase = (size_t)b * SEQ + m0;
    #pragma unroll
    for (int it = 0; it < 8; ++it) {     // 256 x 16B chunks
        int chunk = it * 32 + lane;
        int row = chunk >> 3, off = chunk & 7;
        v8h val = *(const v8h*)(O + row * 64 + off * 8);
        *(v8h*)(Xa + (rowbase + row) * DMODEL + h * DK + off * 8) = val;
    }
}

// ---------------------------------------------------------------------------
// Output projection: out(fp32) = Xa(f16) @ Wo + bo, LDS-staged fp32 stores.
// ---------------------------------------------------------------------------
__global__ void __launch_bounds__(128) k_out(const f16* __restrict__ Xa,
                                             const f16* __restrict__ Wt,
                                             const float* __restrict__ bias,
                                             float* __restrict__ out) {
    __shared__ float cbuf[4][16 * 64];   // 4KB per wave
    int lane = threadIdx.x & 31;
    int wave = threadIdx.x >> 5;
    int t  = blockIdx.x * 4 + wave;
    int tm = t / NHEADS;
    int h  = t - tm * NHEADS;
    int m0 = tm * 16;
    int n0 = h * DK;
    int col = lane & 15, hi = lane >> 4;

    v8f zero = {};
    v8f acc[4] = {zero, zero, zero, zero};
    for (int k0 = 0; k0 < DMODEL; k0 += 32) {
        if (k0 + 64 < DMODEL) {
            __builtin_prefetch(Xa + (size_t)(m0 + col) * DMODEL + k0 + 64 + hi * 8, 0, 3);
            __builtin_prefetch(Wt + (size_t)(n0 + col) * DMODEL + k0 + 64 + hi * 16, 0, 3);
        }
        v16h a = load_a_f16(Xa, DMODEL, m0, k0, lane);
        #pragma unroll
        for (int dt = 0; dt < 4; ++dt) {
            v16h b = load_bt_f16(Wt, DMODEL, n0 + dt * 16, k0, lane);
            acc[dt] = wmma32(a, b, acc[dt]);
        }
    }

    float* C = cbuf[wave];
    #pragma unroll
    for (int dt = 0; dt < 4; ++dt) {
        float bb = bias[n0 + dt * 16 + col];
        #pragma unroll
        for (int r = 0; r < 8; ++r)
            C[(r + 8 * hi) * 64 + dt * 16 + col] = acc[dt][r] + bb;
    }
    STAGE_SYNC();
    #pragma unroll
    for (int it = 0; it < 8; ++it) {     // 256 x 16B chunks
        int chunk = it * 32 + lane;
        int row = chunk >> 4, off = chunk & 15;
        v4f val = *(const v4f*)(C + row * 64 + off * 4);
        *(v4f*)(out + (size_t)(m0 + row) * DMODEL + n0 + off * 4) = val;
    }
}

// ---------------------------------------------------------------------------
extern "C" void kernel_launch(void* const* d_in, const int* in_sizes, int n_in,
                              void* d_out, int out_size, void* d_ws, size_t ws_size,
                              hipStream_t stream) {
    const float* q  = (const float*)d_in[0];
    const float* k  = (const float*)d_in[1];
    const float* v  = (const float*)d_in[2];
    const float* Wq = (const float*)d_in[3];
    const float* bq = (const float*)d_in[4];
    const float* Wk = (const float*)d_in[5];
    const float* bk = (const float*)d_in[6];
    const float* Wv = (const float*)d_in[7];
    const float* bv = (const float*)d_in[8];
    const float* Wo = (const float*)d_in[9];
    const float* bo = (const float*)d_in[10];
    float* out = (float*)d_out;

    char* ws = (char*)d_ws;
    size_t szX = (size_t)MTOT * DMODEL * sizeof(f16);    // 12.58 MB
    size_t szW = (size_t)DMODEL * DMODEL * sizeof(f16);  // 1.18 MB
    f16* Qh  = (f16*)ws; ws += szX;   // [bh][s][64]
    f16* Kh  = (f16*)ws; ws += szX;   // [bh][s][64]
    f16* Vt  = (f16*)ws; ws += szX;   // [bh][64][s]
    f16* Xa  = (f16*)ws; ws += szX;   // [b*s][768]
    f16* Wqt = (f16*)ws; ws += szW;
    f16* Wkt = (f16*)ws; ws += szW;
    f16* Wvt = (f16*)ws; ws += szW;
    f16* Wot = (f16*)ws; ws += szW;
    (void)in_sizes; (void)n_in; (void)out_size; (void)ws_size;

    dim3 blk(128);
    int wtBlocks = (DMODEL * DMODEL + 255) / 256;
    k_wt<<<wtBlocks, 256, 0, stream>>>(Wq, Wqt);
    k_wt<<<wtBlocks, 256, 0, stream>>>(Wk, Wkt);
    k_wt<<<wtBlocks, 256, 0, stream>>>(Wv, Wvt);
    k_wt<<<wtBlocks, 256, 0, stream>>>(Wo, Wot);

    int projBlocks = (MTOT / 16) * NHEADS / 4;           // 1536
    k_proj<0><<<projBlocks, blk, 0, stream>>>(q, Wqt, bq, Qh);
    k_proj<0><<<projBlocks, blk, 0, stream>>>(k, Wkt, bk, Kh);
    k_proj<1><<<projBlocks, blk, 0, stream>>>(v, Wvt, bv, Vt);

    int attnBlocks = (BATCH * NHEADS * (SEQ / 32)) / 4;  // 768
    k_attn<<<attnBlocks, blk, 0, stream>>>(Qh, Kh, Vt, Xa);

    k_out<<<projBlocks, blk, 0, stream>>>(Xa, Wot, bo, out);
}
